// ScaledDotProductAttention_6760278524248
// MI455X (gfx1250) — matrix operand order repaired and checked by reading, not compiled, not run on hardware
//
#include <hip/hip_runtime.h>
#include <hip/hip_bf16.h>
#include <math.h>

// Fused causal attention for MI455X (gfx1250, wave32, WMMA).
// Outputs: attn [B,S,D] then align [B,S,S], concatenated in d_out.
// Roofline: align store (256MB) dominates (~11us @ 23.3TB/s) -> single-pass
// streaming NT stores; Q/K/V (24MB) are L2-resident, so the pass-A score
// recompute is cheap. Pass-A softmax stats are kept per-lane (no per-tile
// cross-lane traffic) with a single-exp online update, merged once at the end.

typedef __attribute__((ext_vector_type(16))) _Float16 v16h;
typedef __attribute__((ext_vector_type(8)))  float    v8f;

#define B_   16
#define S_   2048
#define D_   64
#define QT_  16            // queries per wave (one WMMA M tile)
#define WAVES_ 8           // waves per block
// scores kept in log2 domain: (1/sqrt(64)) * log2(e); v_exp_f32 is exp2
#define SCALE_LOG2 (0.125f * 1.44269504088896340736f)
#define MNEG (-1.0e30f)    // finite "-inf" sentinel for running max

// Raw v_exp_f32 (hardware exp2) — HIP has no __exp2f device intrinsic.
__device__ __forceinline__ float fexp2(float x) {
  return __builtin_amdgcn_exp2f(x);
}

template <bool RA>
__device__ __forceinline__ v8f wmma_f16(v16h a, v16h b, v8f c) {
  // D = A(16x32 f16) * B(32x16 f16) + C(16x16 f32); RA = reuse-A hint
  return __builtin_amdgcn_wmma_f32_16x16x32_f16(
      /*neg_a=*/false, a, /*neg_b=*/false, b,
      /*c_mod=*/(short)0, c, /*reuse_a=*/RA, /*reuse_b=*/false);
}

// A-fragment element mapping (ISA 7.12.2, 16-bit A 16x32):
//   lane l, half h:  M = l&15,  K = 16*(h>>3) + 8*(l>>4) + (h&7)
// A and B as loaded here both use K = 16*(l>>4) + h (same order; B: N = l&15)
// C/D f32 16x16: lane l, vgpr r: N = l&15, M = 8*(l>>4) + r

// Masked score tile (16 queries x 16 keys) in C layout, log2 domain.
__device__ __forceinline__ v8f score_tile(const v16h& a0, const v16h& a1,
                                          const float* __restrict__ Kb,
                                          int kt, int lane, int qbase) {
  const int n  = lane & 15;   // key within tile (B-fragment N)
  const int hi = lane >> 4;
  // Per lane: 16 consecutive d-values of one K row -> coalesced b128 loads.
  const float* kp = Kb + (size_t)(kt * 16 + n) * D_ + (hi << 4);
  v16h b0, b1;
#pragma unroll
  for (int h = 0; h < 16; ++h) {
    b0[h] = (_Float16)kp[h];        // d in [0,32)
    b1[h] = (_Float16)kp[32 + h];   // d in [32,64)
  }
  v8f c = {};
  c = wmma_f16<false>(a0, b0, c);
  c = wmma_f16<false>(a1, b1, c);
  const int mb = hi * 8;
#pragma unroll
  for (int r = 0; r < 8; ++r) {
    float s = c[r] * SCALE_LOG2;
    int gq = qbase + mb + r;
    int gk = kt * 16 + n;
    c[r] = (gk > gq) ? -__builtin_inff() : s;   // causal mask
  }
  return c;
}

__global__ __launch_bounds__(32 * WAVES_)
void attn_fused_kernel(const float* __restrict__ Q, const float* __restrict__ K,
                       const float* __restrict__ V,
                       float* __restrict__ attn, float* __restrict__ align) {
  __shared__ _Float16 ldsP[WAVES_][16][32];   // per-wave P-tile relayout buffer

  const int lane = threadIdx.x & 31;
  const int wave = threadIdx.x >> 5;
  const int qidx = blockIdx.x * WAVES_ + wave;       // global q-tile id
  const int b    = qidx >> 7;                        // S/16 = 128 tiles/batch
  const int qt   = qidx & 127;
  const int qbase = qt * QT_;
  const int gq_last = qbase + QT_ - 1;

  const float* Qb = Q + (size_t)b * S_ * D_;
  const float* Kb = K + (size_t)b * S_ * D_;
  const float* Vb = V + (size_t)b * S_ * D_;

  const int n  = lane & 15;
  const int hi = lane >> 4;
  const int mb = hi * 8;

  // ---- Load Q tile once as two A fragments (d 0..31 / 32..63), f32->f16
  v16h a0, a1;
  {
    const float* qp = Qb + (size_t)(qbase + n) * D_;
#pragma unroll
    for (int h = 0; h < 16; ++h) {
      int k = (hi << 4) + h;
      a0[h] = (_Float16)qp[k];
      a1[h] = (_Float16)qp[32 + k];
    }
  }

  // ---- Pass A: per-LANE online max / sum-exp2 (each lane owns columns
  // k == n (mod 16) of its 8 rows). Single exp per row per tile:
  // mnew = max(m,c); one of {m,c} equals mnew, so exp2 of it is 1.
  float m[8], d[8];
#pragma unroll
  for (int r = 0; r < 8; ++r) { m[r] = MNEG; d[r] = 0.f; }

  for (int kt = 0; kt <= qt; ++kt) {
    if (kt + 1 <= qt)   // prefetch next K tile row (global_prefetch_b8)
      __builtin_prefetch(Kb + (size_t)((kt + 1) * 16 + n) * D_ + (hi << 4), 0, 1);
    v8f c = score_tile(a0, a1, Kb, kt, lane, qbase);
#pragma unroll
    for (int r = 0; r < 8; ++r) {
      float cr   = c[r];                       // may be -inf (masked)
      float mold = m[r];
      float mnew = fmaxf(mold, cr);
      float e    = fexp2(fminf(mold, cr) - mnew);   // exp2(-inf)=0, no NaN
      float keep = d[r] + e;                   // case: cr <= mold
      float resc = __builtin_fmaf(d[r], e, 1.0f);   // case: cr >  mold
      d[r] = (cr <= mold) ? keep : resc;
      m[r] = mnew;
    }
  }

  // ---- Merge (m,d) pairs across the 16-lane half (once, 4 stages)
  float invd[8];
#pragma unroll
  for (int r = 0; r < 8; ++r) {
    float mr = m[r], dr = d[r];
#pragma unroll
    for (int st = 1; st <= 8; st <<= 1) {
      float mo = __shfl_xor(mr, st, 16);
      float dn = __shfl_xor(dr, st, 16);
      float mm = fmaxf(mr, mo);
      dr = dr * fexp2(mr - mm) + dn * fexp2(mo - mm);
      mr = mm;
    }
    m[r]    = mr;             // row max, broadcast across the half
    invd[r] = 1.0f / dr;      // row 1/denominator
  }

  // ---- Pass B: recompute scores, stream normalized probs, accumulate P*V
  v8f o0 = {}, o1 = {}, o2 = {}, o3 = {};
  float* alrow = align + ((size_t)b * S_ + qbase) * S_;

  for (int kb = 0; kb < S_ / 32; ++kb) {
    const int k0 = kb * 32;
    const bool active = (k0 <= gq_last);
    v8f p0 = {}, p1 = {};
    if (active) {
      v8f s0 = score_tile(a0, a1, Kb, 2 * kb,     lane, qbase);
      v8f s1 = score_tile(a0, a1, Kb, 2 * kb + 1, lane, qbase);
#pragma unroll
      for (int r = 0; r < 8; ++r) {
        p0[r] = fexp2(s0[r] - m[r]) * invd[r];   // exp2(-inf)=0 on masked
        p1[r] = fexp2(s1[r] - m[r]) * invd[r];
      }
    }
    // Stream align (write-once 256MB -> non-temporal, keep L2 for K/V).
    // Per r: lanes 0-15 / 16-31 each write one contiguous 64B row segment.
#pragma unroll
    for (int r = 0; r < 8; ++r) {
      float* ar = alrow + (size_t)(mb + r) * S_ + k0 + n;
      __builtin_nontemporal_store(p0[r], ar);
      __builtin_nontemporal_store(p1[r], ar + 16);
    }
    if (active) {
      // Re-layout P (C layout) -> A fragment via per-wave LDS slice.
#pragma unroll
      for (int r = 0; r < 8; ++r) {
        ldsP[wave][mb + r][n]      = (_Float16)p0[r];
        ldsP[wave][mb + r][16 + n] = (_Float16)p1[r];
      }
      v16h pa;
#pragma unroll
      for (int h = 0; h < 16; ++h) {
        int kk = (hi << 4) + h;
        pa[h] = ldsP[wave][n][kk];
      }
      // V block (32 keys x 64 d) as 4 B fragments of 16 columns each.
      // P*V chain shares the A fragment -> set ISA reuse-A hint on 2..4.
      const float* vp = Vb + (size_t)(k0 + (hi << 4)) * D_ + n;
      v16h vb0, vb1, vb2, vb3;
#pragma unroll
      for (int h = 0; h < 16; ++h) {
        const float* row = vp + (size_t)h * D_;
        vb0[h] = (_Float16)row[0];
        vb1[h] = (_Float16)row[16];
        vb2[h] = (_Float16)row[32];
        vb3[h] = (_Float16)row[48];
      }
      o0 = wmma_f16<false>(pa, vb0, o0);
      o1 = wmma_f16<true >(pa, vb1, o1);
      o2 = wmma_f16<true >(pa, vb2, o2);
      o3 = wmma_f16<true >(pa, vb3, o3);
    }
  }

  // ---- Write attn tile (16x64), coalesced 64B per half-wave per row
  float* orow = attn + ((size_t)b * S_ + qbase) * D_;
#pragma unroll
  for (int r = 0; r < 8; ++r) {
    float* op = orow + (size_t)(mb + r) * D_ + n;
    __builtin_nontemporal_store(o0[r], op);
    __builtin_nontemporal_store(o1[r], op + 16);
    __builtin_nontemporal_store(o2[r], op + 32);
    __builtin_nontemporal_store(o3[r], op + 48);
  }
}

extern "C" void kernel_launch(void* const* d_in, const int* in_sizes, int n_in,
                              void* d_out, int out_size, void* d_ws, size_t ws_size,
                              hipStream_t stream) {
  (void)in_sizes; (void)n_in; (void)d_ws; (void)ws_size; (void)out_size;
  const float* q = (const float*)d_in[0];
  const float* k = (const float*)d_in[1];
  const float* v = (const float*)d_in[2];
  // d_in[3] = mask is the analytic causal mask; computed in-kernel.
  float* out   = (float*)d_out;
  float* attn  = out;                              // B*S*D floats
  float* align = out + (size_t)B_ * S_ * D_;       // B*S*S floats
  dim3 grid(B_ * (S_ / QT_) / WAVES_);             // 256 blocks
  dim3 block(32 * WAVES_);                         // 8 waves, one q-tile each
  attn_fused_kernel<<<grid, block, 0, stream>>>(q, k, v, attn, align);
}
